// Attention_69028714382100
// MI455X (gfx1250) — matrix operand order repaired and checked by reading, not compiled, not run on hardware
//
#include <hip/hip_runtime.h>
#include <hip/hip_bf16.h>
#include <math.h>

// ---------------------------------------------------------------------------
// Multi-head attention for MI455X (gfx1250, wave32, WMMA).
// B=8, N=1024, C=768, H=12, HD=64, SCALE=0.125
//
// Pipeline:
//   1) convert x, w_qkv, w_proj to bf16 (workspace)
//   2) QKV GEMM (bf16 WMMA, f32 acc) -> q (pre-scaled), k, v in [B,H,N,64]
//   3) flash attention per (b,h): async K-tile staging to LDS (ASYNCcnt),
//      WMMA QK^T, online softmax, WMMA PV, global_prefetch of next tiles
//   4) proj GEMM (bf16 WMMA, f32 acc) + bias -> f32 output
// Workspace: ~68 MB bf16 staging.
// ---------------------------------------------------------------------------

#define BB 8
#define NN 1024
#define CC 768
#define HH 12
#define HD 64
#define SCALE 0.125f

typedef __bf16 bf16;
typedef __attribute__((ext_vector_type(16))) __bf16 bf16x16;
typedef __attribute__((ext_vector_type(8)))  __bf16 bf16x8;
typedef __attribute__((ext_vector_type(8)))  float  f32x8;
typedef int v4i __attribute__((vector_size(16)));   // matches builtin signature

typedef __attribute__((address_space(1))) v4i* gv4i_p;  // global (AS1)
typedef __attribute__((address_space(3))) v4i* lv4i_p;  // LDS (AS3)

// D = A(16x32) * B(32x16) + C, bf16 in / f32 acc
__device__ __forceinline__ f32x8 wmma_bf16(bf16x16 a, bf16x16 b, f32x8 c) {
  return __builtin_amdgcn_wmma_f32_16x16x32_bf16(false, a, false, b, (short)0, c,
                                                 false, false);
}

// A-fragment (16x32 bf16, row-major source, leading dim ld).
// Layout: lanes 0-15 hold row=lane with K 0..15 contiguous;
//         lanes 16-31 hold row=lane-16 with K 16..31 contiguous.
__device__ __forceinline__ bf16x16 load_frag_a(const bf16* p, int ld, int row_base,
                                               int k_base, int lane) {
  const int r = row_base + (lane & 15);
  const int ksel = (lane >> 4) & 1; // 0 -> K+0, 1 -> K+16
  const bf16* base = p + (size_t)r * ld + k_base + ksel * 16;
  bf16x8 lo = *(const bf16x8*)(base);      // K +0..7   (or +16..23)
  bf16x8 hi = *(const bf16x8*)(base + 8);  // K +8..15  (or +24..31)
  bf16x16 out;
#pragma unroll
  for (int i = 0; i < 8; ++i) { out[i] = lo[i]; out[8 + i] = hi[i]; }
  return out;
}

// B-fragment (32x16 bf16) where B[k][n] = W[n][k], W row-major [Nout, K], ld=K.
// Same contiguous k order as load_frag_a: lanes 0-15 -> n=lane, K 0..15 contiguous;
// lanes 16-31 -> n=lane-16, K 16..31 contiguous.  One 32B contiguous load.
__device__ __forceinline__ bf16x16 load_frag_bT(const bf16* w, int ld, int n_base,
                                                int k_base, int lane) {
  const int n = n_base + (lane & 15);
  const int k0 = k_base + ((lane >> 4) & 1) * 16;
  return *(const bf16x16*)(w + (size_t)n * ld + k0);
}

// ---------------------------------------------------------------------------
// 1) f32 -> bf16 conversion
// ---------------------------------------------------------------------------
__global__ void attn_cvt_bf16(const float* __restrict__ in, bf16* __restrict__ out,
                              int n) {
  int i = blockIdx.x * blockDim.x + threadIdx.x;
  if (i < n) out[i] = (bf16)in[i];
}

// ---------------------------------------------------------------------------
// 2) QKV GEMM: [8192x768] @ [2304x768]^T -> q/k/v [B,H,N,64] bf16, q scaled.
//    Block: 256 thr (8 waves), tile 128(M) x 128(N); wave tile 32x64
//    (8 WMMAs per 6 fragments -> A reuse x4, B reuse x2).
// ---------------------------------------------------------------------------
__global__ __launch_bounds__(256) void attn_qkv_gemm(
    const bf16* __restrict__ A, const bf16* __restrict__ W,
    bf16* __restrict__ qb, bf16* __restrict__ kb, bf16* __restrict__ vb) {
  const int lane = threadIdx.x & 31;
  const int wave = threadIdx.x >> 5;
  const int mbase = blockIdx.x * 128 + (wave & 3) * 32;
  const int nbase = blockIdx.y * 128 + (wave >> 2) * 64;

  f32x8 acc[2][4] = {};
#pragma unroll 2
  for (int k = 0; k < CC; k += 32) {
    bf16x16 afr[2], bfr[4];
#pragma unroll
    for (int mi = 0; mi < 2; ++mi)
      afr[mi] = load_frag_a(A, CC, mbase + mi * 16, k, lane);
#pragma unroll
    for (int ni = 0; ni < 4; ++ni)
      bfr[ni] = load_frag_bT(W, CC, nbase + ni * 16, k, lane);
#pragma unroll
    for (int mi = 0; mi < 2; ++mi)
#pragma unroll
      for (int ni = 0; ni < 4; ++ni)
        acc[mi][ni] = wmma_bf16(afr[mi], bfr[ni], acc[mi][ni]);
  }

  const int half8 = (lane >> 4) * 8;
  const int cl = lane & 15;
#pragma unroll
  for (int mi = 0; mi < 2; ++mi) {
#pragma unroll
    for (int ni = 0; ni < 4; ++ni) {
#pragma unroll
      for (int r = 0; r < 8; ++r) {
        int m = mbase + mi * 16 + r + half8;     // row of X: b*N + n
        int c = nbase + ni * 16 + cl;            // col in [0, 3C)
        int t = c / CC;                          // 0=q 1=k 2=v
        int rem = c - t * CC;
        int h = rem >> 6, d = rem & 63;
        int b = m >> 10, n = m & 1023;
        size_t dst = ((((size_t)b * HH + h) * NN) + n) * HD + d;
        float v = acc[mi][ni][r];
        if (t == 0)      qb[dst] = (bf16)(v * SCALE);
        else if (t == 1) kb[dst] = (bf16)v;
        else             vb[dst] = (bf16)v;
      }
    }
  }
}

// ---------------------------------------------------------------------------
// 3) Flash attention per (b,h).  Block: 128 thr (4 waves) = 64 query rows.
//    Keys streamed in 32-row tiles; K tile staged memory->LDS with CDNA5
//    async copies (ASYNCcnt), V tile transposed manually.
// ---------------------------------------------------------------------------
#define KT_LD 72   // 144B rows (16B multiple)
#define VT_LD 48   // 96B rows  (16B multiple)

__global__ __launch_bounds__(128) void attn_core(
    const bf16* __restrict__ qg, const bf16* __restrict__ kg,
    const bf16* __restrict__ vg, bf16* __restrict__ aout) {
  __shared__ bf16 Kt[32 * KT_LD];      // [key][d]
  __shared__ bf16 Vt[64 * VT_LD];      // [d][key]   (transposed)
  __shared__ bf16 Pst[4][16 * 32];     // per-wave P tile, A-fragment source

  const int tid = threadIdx.x;
  const int lane = tid & 31;
  const int wave = tid >> 5;
  const int bh = blockIdx.y;           // b*H + h
  const int b = bh / HH, h = bh - b * HH;
  const int qbase = blockIdx.x * 64 + wave * 16;   // query row in [0,N)

  const bf16* Q = qg + (size_t)bh * NN * HD;
  const bf16* K = kg + (size_t)bh * NN * HD;
  const bf16* V = vg + (size_t)bh * NN * HD;

  // Q fragments for this wave's 16 rows (K-dim 0..31 and 32..63), loaded once.
  bf16x16 aq0 = load_frag_a(Q, HD, qbase, 0, lane);
  bf16x16 aq1 = load_frag_a(Q, HD, qbase, 32, lane);

  const int half8 = (lane >> 4) * 8;
  const int cl = lane & 15;
  bf16* Pw = &Pst[wave][0];

  float mrow[8], lrow[8], arow[8];
  f32x8 o0 = {}, o1 = {}, o2 = {}, o3 = {};
#pragma unroll
  for (int r = 0; r < 8; ++r) { mrow[r] = -1.0e30f; lrow[r] = 0.0f; }

#pragma unroll 2
  for (int kb = 0; kb < NN; kb += 32) {
    __syncthreads();   // previous tile fully consumed before overwrite

    // --- K tile (32x64 bf16 = 4KB): async memory->LDS, 16B per op, 2 per thr
    {
      int c = tid;
#pragma unroll
      for (int it = 0; it < 2; ++it, c += 128) {
        int key = c >> 3, d8 = (c & 7) * 8;
        __builtin_amdgcn_global_load_async_to_lds_b128(
            (gv4i_p)(const void*)(K + (size_t)(kb + key) * HD + d8),
            (lv4i_p)(void*)(&Kt[key * KT_LD + d8]),
            0, 0);
      }
    }
    // --- V tile transposed into LDS (manual: async copy cannot transpose)
    for (int c = tid; c < 32 * 8; c += 128) {
      int key = c >> 3, d8 = (c & 7) * 8;
      bf16x8 vv = *(const bf16x8*)(V + (size_t)(kb + key) * HD + d8);
#pragma unroll
      for (int i = 0; i < 8; ++i) Vt[(d8 + i) * VT_LD + key] = vv[i];
    }
    // --- prefetch next tiles into cache while this one is consumed
    if (kb + 32 < NN) {
      __builtin_prefetch((const void*)(K + (size_t)(kb + 32) * HD + (tid & 127) * 16), 0, 1);
      __builtin_prefetch((const void*)(V + (size_t)(kb + 32) * HD + (tid & 127) * 16), 0, 1);
    }
    asm volatile("s_wait_asynccnt 0x0" ::: "memory");  // own async copies done
    __syncthreads();                                   // whole tile visible

    // S = Q(16x64) @ K^T(64x{16,16}) : B frag lane -> key=lane&15 (row of Kt),
    // d contiguous per half-wave.
    f32x8 s0 = {}, s1 = {};
    {
      const int dc = ((lane >> 4) & 1) * 16;
      bf16x16 bk00 = *(const bf16x16*)(&Kt[(0 + cl) * KT_LD + 0 + dc]);
      bf16x16 bk01 = *(const bf16x16*)(&Kt[(0 + cl) * KT_LD + 32 + dc]);
      bf16x16 bk10 = *(const bf16x16*)(&Kt[(16 + cl) * KT_LD + 0 + dc]);
      bf16x16 bk11 = *(const bf16x16*)(&Kt[(16 + cl) * KT_LD + 32 + dc]);
      s0 = wmma_bf16(aq0, bk00, s0);
      s0 = wmma_bf16(aq1, bk01, s0);
      s1 = wmma_bf16(aq0, bk10, s1);
      s1 = wmma_bf16(aq1, bk11, s1);
    }

    // Online softmax (rows live in VGPR index r, replicated over 16 lanes/half)
#pragma unroll
    for (int r = 0; r < 8; ++r) {
      float mx = fmaxf(s0[r], s1[r]);
#pragma unroll
      for (int msk = 8; msk >= 1; msk >>= 1)
        mx = fmaxf(mx, __shfl_xor(mx, msk, 32));
      float mnew = fmaxf(mrow[r], mx);
      float alpha = __expf(mrow[r] - mnew);
      float p0 = __expf(s0[r] - mnew);
      float p1 = __expf(s1[r] - mnew);
      float ps = p0 + p1;
#pragma unroll
      for (int msk = 8; msk >= 1; msk >>= 1)
        ps += __shfl_xor(ps, msk, 32);
      lrow[r] = lrow[r] * alpha + ps;
      mrow[r] = mnew;
      arow[r] = alpha;
      int row = r + half8;
      Pw[row * 32 + cl] = (bf16)p0;        // keys kb..kb+15
      Pw[row * 32 + 16 + cl] = (bf16)p1;   // keys kb+16..kb+31
    }

    // rescale running O by alpha
#pragma unroll
    for (int r = 0; r < 8; ++r) {
      o0[r] *= arow[r]; o1[r] *= arow[r]; o2[r] *= arow[r]; o3[r] *= arow[r];
    }

    // O += P(16x32) @ V(32x64): A frag from LDS P tile, B frags from Vt rows.
    bf16x16 ap = load_frag_a(Pw, 32, 0, 0, lane);
    {
      const int k0 = ((lane >> 4) & 1) * 16;
      bf16x16 bv0 = *(const bf16x16*)(&Vt[(0 + cl) * VT_LD + k0]);
      bf16x16 bv1 = *(const bf16x16*)(&Vt[(16 + cl) * VT_LD + k0]);
      bf16x16 bv2 = *(const bf16x16*)(&Vt[(32 + cl) * VT_LD + k0]);
      bf16x16 bv3 = *(const bf16x16*)(&Vt[(48 + cl) * VT_LD + k0]);
      o0 = wmma_bf16(ap, bv0, o0);
      o1 = wmma_bf16(ap, bv1, o1);
      o2 = wmma_bf16(ap, bv2, o2);
      o3 = wmma_bf16(ap, bv3, o3);
    }
  }

  // normalize and write [B,N,C] bf16 (row-major, head-interleaved columns)
#pragma unroll
  for (int r = 0; r < 8; ++r) {
    float inv = 1.0f / lrow[r];
    int row_g = qbase + r + half8;                 // n index
    size_t base = ((size_t)b * NN + row_g) * CC + h * HD;
    aout[base + 0 * 16 + cl]  = (bf16)(o0[r] * inv);
    aout[base + 1 * 16 + cl]  = (bf16)(o1[r] * inv);
    aout[base + 2 * 16 + cl]  = (bf16)(o2[r] * inv);
    aout[base + 3 * 16 + cl]  = (bf16)(o3[r] * inv);
  }
}

// ---------------------------------------------------------------------------
// 4) Proj GEMM: [8192x768] @ [768x768]^T + bias -> f32 out
//    Same 128x128 block / 32x64 wave tiling as QKV GEMM.
// ---------------------------------------------------------------------------
__global__ __launch_bounds__(256) void attn_proj_gemm(
    const bf16* __restrict__ A, const bf16* __restrict__ W,
    const float* __restrict__ bias, float* __restrict__ out) {
  const int lane = threadIdx.x & 31;
  const int wave = threadIdx.x >> 5;
  const int mbase = blockIdx.x * 128 + (wave & 3) * 32;
  const int nbase = blockIdx.y * 128 + (wave >> 2) * 64;

  f32x8 acc[2][4] = {};
#pragma unroll 2
  for (int k = 0; k < CC; k += 32) {
    bf16x16 afr[2], bfr[4];
#pragma unroll
    for (int mi = 0; mi < 2; ++mi)
      afr[mi] = load_frag_a(A, CC, mbase + mi * 16, k, lane);
#pragma unroll
    for (int ni = 0; ni < 4; ++ni)
      bfr[ni] = load_frag_bT(W, CC, nbase + ni * 16, k, lane);
#pragma unroll
    for (int mi = 0; mi < 2; ++mi)
#pragma unroll
      for (int ni = 0; ni < 4; ++ni)
        acc[mi][ni] = wmma_bf16(afr[mi], bfr[ni], acc[mi][ni]);
  }

  const int half8 = (lane >> 4) * 8;
  const int cl = lane & 15;
#pragma unroll
  for (int mi = 0; mi < 2; ++mi) {
#pragma unroll
    for (int ni = 0; ni < 4; ++ni) {
      int c = nbase + ni * 16 + cl;
      float bv = bias[c];
#pragma unroll
      for (int r = 0; r < 8; ++r) {
        int m = mbase + mi * 16 + r + half8;
        out[(size_t)m * CC + c] = acc[mi][ni][r] + bv;
      }
    }
  }
}

// ---------------------------------------------------------------------------
// launch
// ---------------------------------------------------------------------------
extern "C" void kernel_launch(void* const* d_in, const int* in_sizes, int n_in,
                              void* d_out, int out_size, void* d_ws, size_t ws_size,
                              hipStream_t stream) {
  const float* x      = (const float*)d_in[0];  // [B,N,C]
  const float* w_qkv  = (const float*)d_in[1];  // [3C,C]
  const float* w_proj = (const float*)d_in[2];  // [C,C]
  const float* b_proj = (const float*)d_in[3];  // [C]
  float* out = (float*)d_out;

  char* ws = (char*)d_ws;
  size_t off = 0;
  auto carve = [&](size_t bytes) -> void* {
    void* p = ws + off;
    off = (off + bytes + 255) & ~(size_t)255;
    return p;
  };
  const size_t n_x   = (size_t)BB * NN * CC;          // 6.29M
  const size_t n_qkv = (size_t)3 * CC * CC;           // 1.77M
  const size_t n_pj  = (size_t)CC * CC;               // 0.59M
  const size_t n_hd  = (size_t)BB * HH * NN * HD;     // 6.29M

  bf16* xb     = (bf16*)carve(n_x * 2);
  bf16* wqkvb  = (bf16*)carve(n_qkv * 2);
  bf16* wprojb = (bf16*)carve(n_pj * 2);
  bf16* qb     = (bf16*)carve(n_hd * 2);
  bf16* kb     = (bf16*)carve(n_hd * 2);
  bf16* vb     = (bf16*)carve(n_hd * 2);
  bf16* aout   = (bf16*)carve(n_x * 2);
  (void)ws_size; // total ~68 MB

  attn_cvt_bf16<<<(int)((n_x + 255) / 256), 256, 0, stream>>>(x, xb, (int)n_x);
  attn_cvt_bf16<<<(int)((n_qkv + 255) / 256), 256, 0, stream>>>(w_qkv, wqkvb, (int)n_qkv);
  attn_cvt_bf16<<<(int)((n_pj + 255) / 256), 256, 0, stream>>>(w_proj, wprojb, (int)n_pj);

  dim3 gq(BB * NN / 128, 3 * CC / 128);          // (64, 18)
  attn_qkv_gemm<<<gq, 256, 0, stream>>>(xb, wqkvb, qb, kb, vb);

  dim3 ga(NN / 64, BB * HH);                     // (16, 96)
  attn_core<<<ga, 128, 0, stream>>>(qb, kb, vb, aout);

  dim3 gp(BB * NN / 128, CC / 128);              // (64, 6)
  attn_proj_gemm<<<gp, 256, 0, stream>>>(aout, wprojb, b_proj, out);
}
